// FactorAtt_ConvRelPosEnc_SE1adapt_37142877176321
// MI455X (gfx1250) — matrix operand order repaired and checked by reading, not compiled, not run on hardware
//
#include <hip/hip_runtime.h>

// FactorAtt + ConvRelPosEnc + SE adapter for MI455X (gfx1250), wave32 + WMMA bf16.
// B=8, N=4096 (64x64), DIM=512, HEADS=8, CH=64, HIDDEN=256.
// Workspace (~169.4 MB): qb/kb/vb/convv/fab bf16 [32768,512], kmax/ksum f32 [8,512],
// ktv_p f32 [64,64,64], ktvb bf16 [64,64,64], pooled f32 [8,512], gatep f32 [8,8].

typedef __bf16 bf16_t;
typedef __attribute__((ext_vector_type(16))) __bf16 v16bf;
typedef __attribute__((ext_vector_type(4)))  __bf16 bf16x4;
typedef __attribute__((ext_vector_type(8)))  float  v8f;
typedef __attribute__((ext_vector_type(8)))  int    v8i;

#define TS 40  // padded LDS stride in bf16 elements (32 data + 8 pad) -> 80B rows, 16B-aligned

__device__ inline v8f wmma_bf16(v16bf a, v16bf b, v8f c) {
  // D = A(16x32) * B(32x16) + C, f32 accumulate
  return __builtin_amdgcn_wmma_f32_16x16x32_bf16(false, a, false, b, (short)0, c, false, false);
}

// A fragment: lds layout [row][k] (stride=TS). Lane L: M=L&15; K pairs per ISA table.
__device__ inline v16bf load_frag_a(const bf16_t* lds, int rowBase) {
  const int lane = threadIdx.x & 31;
  const int m = lane & 15;
  const int hi8 = (lane & 16) ? 8 : 0;
  const bf16_t* p = lds + (rowBase + m) * TS;
  v8i raw;
#pragma unroll
  for (int j = 0; j < 8; ++j) {
    int k = 2 * j + ((j >= 4) ? 8 : 0) + hi8;   // {0,2,4,6,16,18,20,22} + laneHi*8
    raw[j] = *(const int*)(p + k);
  }
  return __builtin_bit_cast(v16bf, raw);
}

// B fragment: lds layout [col][k] (staged transposed). Lane L: N=L&15;
// VGPR j holds rows K = (L>=16?8:0) + {2j, 2j+1} + (j>=4?8:0).
__device__ inline v16bf load_frag_b(const bf16_t* lds, int colBase) {
  const int lane = threadIdx.x & 31;
  const int c = lane & 15;
  const int kb0 = (lane & 16) ? 8 : 0;
  const bf16_t* p = lds + (colBase + c) * TS + kb0;
  v8i raw;
#pragma unroll
  for (int j = 0; j < 8; ++j) raw[j] = *(const int*)(p + 2 * j + ((j >= 4) ? 8 : 0));
  return __builtin_bit_cast(v16bf, raw);
}

// Async 16B copy global->LDS (CDNA5 TDM-lite path, tracked by ASYNCcnt).
// lds_addr: wave-relative LDS byte address = low 32 bits of generic shared pointer.
__device__ inline void async_copy_b128(unsigned lds_addr, const bf16_t* gsrc) {
  asm volatile("global_load_async_to_lds_b128 %0, %1, off"
               :: "v"(lds_addr), "v"((unsigned long long)(uintptr_t)gsrc)
               : "memory");
}
__device__ inline void wait_async0() {
  asm volatile("s_wait_asynccnt 0" ::: "memory");
}

// ---------------- QKV GEMM: [32768,512]f32 @ [512,1536]f32 -> q/k/v bf16 ----------------
__global__ __launch_bounds__(256) void qkv_gemm_kernel(
    const float* __restrict__ x, const float* __restrict__ wqkv,
    bf16_t* __restrict__ qb, bf16_t* __restrict__ kb, bf16_t* __restrict__ vb) {
  __shared__ bf16_t ldsA[128 * TS];
  __shared__ bf16_t ldsB[128 * TS];
  const int N0 = blockIdx.x * 128;        // 0..1535
  const int M0 = blockIdx.y * 128;        // 0..32767
  const int tid = threadIdx.x, wave = tid >> 5, lane = tid & 31;
  v8f acc[8] = {};
  for (int kt = 0; kt < 16; ++kt) {
    const int kg = kt * 32;
    for (int i = tid; i < 128 * 8; i += 256) {       // A: f32x4 -> bf16x4
      int r = i >> 3, c4 = (i & 7) * 4;
      float4 f = *(const float4*)&x[(size_t)(M0 + r) * 512 + kg + c4];
      bf16x4 o = {(bf16_t)f.x, (bf16_t)f.y, (bf16_t)f.z, (bf16_t)f.w};
      *(bf16x4*)&ldsA[r * TS + c4] = o;
    }
    for (int i = tid; i < 128 * 8; i += 256) {       // B transposed: [col][k], 4 k per thread
      int c = i & 127, k0 = (i >> 7) * 4;
      const float* wp = &wqkv[(size_t)(kg + k0) * 1536 + N0 + c];
      bf16x4 o = {(bf16_t)wp[0], (bf16_t)wp[1536], (bf16_t)wp[2 * 1536], (bf16_t)wp[3 * 1536]};
      *(bf16x4*)&ldsB[c * TS + k0] = o;
    }
    __syncthreads();
    v16bf a = load_frag_a(ldsA, wave * 16);
#pragma unroll
    for (int ct = 0; ct < 8; ++ct) {
      v16bf b = load_frag_b(ldsB, ct * 16);
      acc[ct] = wmma_bf16(a, b, acc[ct]);
    }
    __syncthreads();
  }
  const int cl = lane & 15, hi8 = (lane & 16) ? 8 : 0;
#pragma unroll
  for (int ct = 0; ct < 8; ++ct)
#pragma unroll
    for (int r = 0; r < 8; ++r) {
      int row = M0 + wave * 16 + r + hi8;
      int col = N0 + ct * 16 + cl;                   // whole block within one 512-segment
      bf16_t v = (bf16_t)acc[ct][r];
      if (col < 512)       qb[(size_t)row * 512 + col]          = v;
      else if (col < 1024) kb[(size_t)row * 512 + (col - 512)]  = v;
      else                 vb[(size_t)row * 512 + (col - 1024)] = v;
    }
}

// ---------------- Depthwise conv (CRPE), SAME padding, 8x8 pixel tile + halo ----------------
template <int KS>
__device__ inline void conv_body(const bf16_t* ldsV, const float* wp, int tid, int cc,
                                 int b, int ty0, int tx0, int c, bf16_t* __restrict__ convv) {
  constexpr int R = KS / 2;
  float wreg[KS * KS];
#pragma unroll
  for (int i = 0; i < KS * KS; ++i) wreg[i] = wp[i];
#pragma unroll
  for (int o = 0; o < 16; ++o) {
    int pidx = (tid >> 6) + o * 4;                   // 0..63
    int py = pidx >> 3, px = pidx & 7;
    float acc = 0.f;
#pragma unroll
    for (int dy = 0; dy < KS; ++dy)
#pragma unroll
      for (int dx = 0; dx < KS; ++dx)
        acc += (float)ldsV[((py + 3 - R + dy) * 14 + (px + 3 - R + dx)) * 64 + cc] *
               wreg[dy * KS + dx];
    convv[((size_t)(b * 4096 + (ty0 + py) * 64 + (tx0 + px))) * 512 + c] = (bf16_t)acc;
  }
}

__global__ __launch_bounds__(256) void dwconv_kernel(
    const bf16_t* __restrict__ vb, const float* __restrict__ w3,
    const float* __restrict__ w5, const float* __restrict__ w7,
    bf16_t* __restrict__ convv) {
  __shared__ bf16_t ldsV[14 * 14 * 64];              // halo (r=3) x 64 channels
  const int b = blockIdx.x >> 6, tile = blockIdx.x & 63;
  const int chunk = blockIdx.y;                      // 64-channel chunk, 0..7
  const int ty0 = (tile >> 3) * 8, tx0 = (tile & 7) * 8;
  const int tid = threadIdx.x;
  for (int i = tid; i < 196 * 8; i += 256) {         // 16B chunks
    int pix = i >> 3, s8 = (i & 7) * 8;
    int gy = ty0 - 3 + pix / 14, gx = tx0 - 3 + pix % 14;
    uint4 val = {0u, 0u, 0u, 0u};
    if (gy >= 0 && gy < 64 && gx >= 0 && gx < 64)
      val = *(const uint4*)&vb[((size_t)(b * 4096 + gy * 64 + gx)) * 512 + chunk * 64 + s8];
    *(uint4*)&ldsV[pix * 64 + s8] = val;
  }
  __syncthreads();
  const int cc = tid & 63;
  const int c = chunk * 64 + cc;
  if (chunk < 2)      conv_body<3>(ldsV, w3 + c * 9,          tid, cc, b, ty0, tx0, c, convv);
  else if (chunk < 5) conv_body<5>(ldsV, w5 + (c - 128) * 25, tid, cc, b, ty0, tx0, c, convv);
  else                conv_body<7>(ldsV, w7 + (c - 320) * 49, tid, cc, b, ty0, tx0, c, convv);
}

// ---------------- softmax stats over N per (b, h*64+ch): online max + sum(exp) ----------------
__global__ __launch_bounds__(256) void kstats_kernel(
    const bf16_t* __restrict__ kb, float* __restrict__ kmax, float* __restrict__ ksum) {
  __shared__ float lm[256], ls[256];
  const int cg = blockIdx.x, b = blockIdx.y;
  const int tid = threadIdx.x, cc = tid & 63, rr = tid >> 6;
  const int c = cg * 64 + cc;
  float m = -3.0e38f, s = 0.f;
  for (int n = rr; n < 4096; n += 4) {
    float xv = (float)kb[((size_t)(b * 4096 + n)) * 512 + c];
    if (xv > m) { s = s * __expf(m - xv) + 1.f; m = xv; }
    else        { s += __expf(xv - m); }
  }
  lm[tid] = m; ls[tid] = s;
  __syncthreads();
  if (rr == 0) {
    float M = m, S = s;
    for (int j = 1; j < 4; ++j) {
      float mj = lm[j * 64 + cc], sj = ls[j * 64 + cc];
      float Mn = fmaxf(M, mj);
      S = S * __expf(M - Mn) + sj * __expf(mj - Mn);
      M = Mn;
    }
    kmax[b * 512 + c] = M;
    ksum[b * 512 + c] = S;
  }
}

__global__ void zero_kernel(float* __restrict__ p, int n) {
  int i = blockIdx.x * 256 + threadIdx.x;
  if (i < n) p[i] = 0.f;
}

// ---------------- ktv = exp(k-max)^T @ v per (b,h): WMMA, K split 8 ways ----------------
__global__ __launch_bounds__(128) void ktv_kernel(
    const bf16_t* __restrict__ kb, const bf16_t* __restrict__ vb,
    const float* __restrict__ kmax, float* __restrict__ ktv_part) {
  __shared__ bf16_t ldsK[64 * TS];                   // [ck][n] (transposed exp(k))
  __shared__ bf16_t ldsV[64 * TS];                   // [cv][n] (transposed v)
  const int ks = blockIdx.x;                         // K split 0..7 (512 tokens each)
  const int bh = blockIdx.y;                         // 0..63
  const int b = bh >> 3, hc0 = (bh & 7) * 64;
  const int tid = threadIdx.x, wave = tid >> 5, lane = tid & 31;
  v8f acc[4] = {};
  const int n0 = ks * 512;
  for (int kt = 0; kt < 16; ++kt) {
    const int nb = n0 + kt * 32;
    for (int i = tid; i < 16 * 32; i += 128) {       // 4 channels per thread
      int c4 = (i & 15) * 4, n = i >> 4;
      size_t g = ((size_t)(b * 4096 + nb + n)) * 512 + hc0 + c4;
      bf16x4 kv = *(const bf16x4*)&kb[g];
      bf16x4 vv = *(const bf16x4*)&vb[g];
      float4 km = *(const float4*)&kmax[b * 512 + hc0 + c4];
      ldsK[(c4 + 0) * TS + n] = (bf16_t)__expf((float)kv[0] - km.x);
      ldsK[(c4 + 1) * TS + n] = (bf16_t)__expf((float)kv[1] - km.y);
      ldsK[(c4 + 2) * TS + n] = (bf16_t)__expf((float)kv[2] - km.z);
      ldsK[(c4 + 3) * TS + n] = (bf16_t)__expf((float)kv[3] - km.w);
      ldsV[(c4 + 0) * TS + n] = vv[0];
      ldsV[(c4 + 1) * TS + n] = vv[1];
      ldsV[(c4 + 2) * TS + n] = vv[2];
      ldsV[(c4 + 3) * TS + n] = vv[3];
    }
    __syncthreads();
    v16bf a = load_frag_a(ldsK, wave * 16);
#pragma unroll
    for (int ct = 0; ct < 4; ++ct) {
      v16bf bf = load_frag_b(ldsV, ct * 16);
      acc[ct] = wmma_bf16(a, bf, acc[ct]);
    }
    __syncthreads();
  }
  const int cl = lane & 15, hi8 = (lane & 16) ? 8 : 0;
#pragma unroll
  for (int ct = 0; ct < 4; ++ct)
#pragma unroll
    for (int r = 0; r < 8; ++r) {
      int ck = wave * 16 + r + hi8;
      int cv = ct * 16 + cl;
      atomicAdd(&ktv_part[((size_t)bh * 64 + ck) * 64 + cv], acc[ct][r]);
    }
}

// finalize: /sum(exp), * CH^-0.5 (=0.125), -> bf16
__global__ void ktvfin_kernel(const float* __restrict__ part, const float* __restrict__ ksum,
                              bf16_t* __restrict__ ktvb) {
  int i = blockIdx.x * 256 + threadIdx.x;            // 0..262143
  int ck = (i >> 6) & 63;
  int bh = i >> 12;
  int b = bh >> 3, h = bh & 7;
  float s = ksum[b * 512 + h * 64 + ck];
  ktvb[i] = (bf16_t)(part[i] / s * 0.125f);
}

// ---------------- factor_att = q@ktv + q*conv_v ; accumulate pooled sums ----------------
__global__ __launch_bounds__(256) void fa_kernel(
    const bf16_t* __restrict__ qb, const bf16_t* __restrict__ ktvb,
    const bf16_t* __restrict__ convv, bf16_t* __restrict__ fab,
    float* __restrict__ pooled) {
  __shared__ bf16_t ldsA[128 * TS];
  __shared__ bf16_t ldsB[64 * TS];
  __shared__ float  ldsP[64];
  const int n0 = blockIdx.x * 128;
  const int h = blockIdx.y, b = blockIdx.z;
  const int hc0 = h * 64, bh = b * 8 + h;
  const int tid = threadIdx.x, wave = tid >> 5, lane = tid & 31;
  const unsigned ldsA_base = (unsigned)(uintptr_t)(void*)&ldsA[0];
  v8f acc[4] = {};
  for (int kt = 0; kt < 2; ++kt) {
    const int kbase = kt * 32;
    for (int i = tid; i < 128 * 4; i += 256) {       // A: pure bf16 copy -> async to LDS
      int r = i >> 2, c8 = (i & 3) * 8;              // 16B chunks
      async_copy_b128(ldsA_base + (unsigned)((r * TS + c8) * 2),
                      &qb[((size_t)(b * 4096 + n0 + r)) * 512 + hc0 + kbase + c8]);
    }
    for (int i = tid; i < 16 * 32; i += 256) {       // B transposed: [cv][ck]
      int cv4 = (i & 15) * 4, k = i >> 4;
      bf16x4 t = *(const bf16x4*)&ktvb[((size_t)bh * 64 + kbase + k) * 64 + cv4];
      ldsB[(cv4 + 0) * TS + k] = t[0];
      ldsB[(cv4 + 1) * TS + k] = t[1];
      ldsB[(cv4 + 2) * TS + k] = t[2];
      ldsB[(cv4 + 3) * TS + k] = t[3];
    }
    wait_async0();
    __syncthreads();
    v16bf a = load_frag_a(ldsA, wave * 16);
#pragma unroll
    for (int ct = 0; ct < 4; ++ct) {
      v16bf bb = load_frag_b(ldsB, ct * 16);
      acc[ct] = wmma_bf16(a, bb, acc[ct]);
    }
    __syncthreads();
  }
  if (tid < 64) ldsP[tid] = 0.f;
  __syncthreads();
  const int cl = lane & 15, hi8 = (lane & 16) ? 8 : 0;
#pragma unroll
  for (int ct = 0; ct < 4; ++ct)
#pragma unroll
    for (int r = 0; r < 8; ++r) {
      int n = n0 + wave * 16 + r + hi8;
      int ch = ct * 16 + cl;
      size_t g = ((size_t)(b * 4096 + n)) * 512 + hc0 + ch;
      float fa = acc[ct][r] + (float)qb[g] * (float)convv[g];   // + CRPE
      fab[g] = (bf16_t)fa;
      atomicAdd(&ldsP[ch], fa);
    }
  __syncthreads();
  if (tid < 64) atomicAdd(&pooled[b * 512 + hc0 + tid], ldsP[tid]);
}

// ---------------- SE gate: 1 + sigmoid(relu(mean @ W_t1^T) @ W_t2^T) ----------------
__global__ __launch_bounds__(256) void gate_kernel(
    const float* __restrict__ pooled, const float* __restrict__ wt1,
    const float* __restrict__ wt2, float* __restrict__ gatep) {
  __shared__ float lp[512];
  __shared__ float lh[256];
  const int b = blockIdx.x, tid = threadIdx.x;
  for (int i = tid; i < 512; i += 256) lp[i] = pooled[b * 512 + i] * (1.f / 4096.f);
  __syncthreads();
  float s = 0.f;
  for (int c = 0; c < 512; ++c) s += lp[c] * wt1[tid * 512 + c];
  lh[tid] = fmaxf(s, 0.f);
  __syncthreads();
  if (tid < 8) {
    float g = 0.f;
    for (int j = 0; j < 256; ++j) g += lh[j] * wt2[tid * 256 + j];
    gatep[b * 8 + tid] = 1.f + 1.f / (1.f + __expf(-g));
  }
}

// ---------------- projection: out = fa @ (gate-scaled Wproj^T) + b_proj ----------------
// (1+gate)[b, k>>6] folded into B staging: out = sum_k A[r,k] * g[k] * B[k,o].
__global__ __launch_bounds__(256) void proj_kernel(
    const bf16_t* __restrict__ fab, const float* __restrict__ wproj,
    const float* __restrict__ bproj, const float* __restrict__ gatep,
    float* __restrict__ out) {
  __shared__ bf16_t ldsA[128 * TS];
  __shared__ bf16_t ldsB[128 * TS];
  const int N0 = blockIdx.x * 128;
  const int M0 = blockIdx.y * 128;
  const int b = M0 >> 12;                            // 4096 rows per batch
  const int tid = threadIdx.x, wave = tid >> 5, lane = tid & 31;
  const unsigned ldsA_base = (unsigned)(uintptr_t)(void*)&ldsA[0];
  v8f acc[8] = {};
  for (int kt = 0; kt < 16; ++kt) {
    const int kg = kt * 32;
    for (int i = tid; i < 128 * 4; i += 256) {       // A: pure bf16 copy -> async to LDS
      int r = i >> 2, c8 = (i & 3) * 8;
      async_copy_b128(ldsA_base + (unsigned)((r * TS + c8) * 2),
                      &fab[((size_t)(M0 + r)) * 512 + kg + c8]);
    }
    for (int i = tid; i < 128 * 8; i += 256) {       // B = Wproj^T * (1+gate): [c_out][k]
      int c = i & 127, k0 = (i >> 7) * 4;
      float4 f = *(const float4*)&wproj[(size_t)(N0 + c) * 512 + kg + k0];
      float g = gatep[b * 8 + ((kg + k0) >> 6)];     // head uniform across the 4 k's
      bf16x4 o = {(bf16_t)(f.x * g), (bf16_t)(f.y * g), (bf16_t)(f.z * g), (bf16_t)(f.w * g)};
      *(bf16x4*)&ldsB[c * TS + k0] = o;
    }
    wait_async0();
    __syncthreads();
    v16bf a = load_frag_a(ldsA, wave * 16);
#pragma unroll
    for (int ct = 0; ct < 8; ++ct) {
      v16bf bb = load_frag_b(ldsB, ct * 16);
      acc[ct] = wmma_bf16(a, bb, acc[ct]);
    }
    __syncthreads();
  }
  const int cl = lane & 15, hi8 = (lane & 16) ? 8 : 0;
#pragma unroll
  for (int ct = 0; ct < 8; ++ct)
#pragma unroll
    for (int r = 0; r < 8; ++r) {
      int row = M0 + wave * 16 + r + hi8;
      int col = N0 + ct * 16 + cl;
      out[(size_t)row * 512 + col] = acc[ct][r] + bproj[col];
    }
}

extern "C" void kernel_launch(void* const* d_in, const int* in_sizes, int n_in,
                              void* d_out, int out_size, void* d_ws, size_t ws_size,
                              hipStream_t stream) {
  (void)in_sizes; (void)n_in; (void)out_size; (void)ws_size;
  const float* x     = (const float*)d_in[0];
  const float* wqkv  = (const float*)d_in[1];
  const float* wproj = (const float*)d_in[2];
  const float* bproj = (const float*)d_in[3];
  const float* wt1   = (const float*)d_in[4];
  const float* wt2   = (const float*)d_in[5];
  const float* w3    = (const float*)d_in[6];
  const float* w5    = (const float*)d_in[7];
  const float* w7    = (const float*)d_in[8];
  // d_in[9], d_in[10] are H=64, W=64 (compile-time constants here)

  char* ws = (char*)d_ws;
  const size_t SZ = (size_t)32768 * 512 * 2;         // one bf16 [B*N, 512] buffer
  bf16_t* qb     = (bf16_t*)(ws);
  bf16_t* kb     = (bf16_t*)(ws + 1 * SZ);
  bf16_t* vb     = (bf16_t*)(ws + 2 * SZ);
  bf16_t* convv  = (bf16_t*)(ws + 3 * SZ);
  bf16_t* fab    = (bf16_t*)(ws + 4 * SZ);
  float*  kmax   = (float*)(ws + 5 * SZ);
  float*  ksum   = kmax + 4096;
  float*  ktv_p  = ksum + 4096;                      // 64*64*64 f32
  bf16_t* ktvb   = (bf16_t*)(ktv_p + 262144);
  float*  pooled = (float*)((char*)ktvb + 262144 * 2);
  float*  gatep  = pooled + 4096;

  zero_kernel<<<dim3(1024), 256, 0, stream>>>(ktv_p, 262144);
  zero_kernel<<<dim3(16),   256, 0, stream>>>(pooled, 4096);
  qkv_gemm_kernel<<<dim3(12, 256), 256, 0, stream>>>(x, wqkv, qb, kb, vb);
  dwconv_kernel<<<dim3(512, 8), 256, 0, stream>>>(vb, w3, w5, w7, convv);
  kstats_kernel<<<dim3(8, 8), 256, 0, stream>>>(kb, kmax, ksum);
  ktv_kernel<<<dim3(8, 64), 128, 0, stream>>>(kb, vb, kmax, ktv_p);
  ktvfin_kernel<<<dim3(1024), 256, 0, stream>>>(ktv_p, ksum, ktvb);
  fa_kernel<<<dim3(32, 8, 8), 256, 0, stream>>>(qb, ktvb, convv, fab, pooled);
  gate_kernel<<<dim3(8), 256, 0, stream>>>(pooled, wt1, wt2, gatep);
  proj_kernel<<<dim3(4, 256), 256, 0, stream>>>(fab, wproj, bproj, gatep, (float*)d_out);
}
